// FPWithLSTM_40767829573879
// MI455X (gfx1250) — compile-verified, matrix operand order checked by reading, not run on hardware
//
#include <hip/hip_runtime.h>
#include <hip/hip_bf16.h>
#include <math.h>

// ---------------- problem constants ----------------
#define BB   128
#define SS   256
#define HH   256
#define EE   256
#define NN   (BB * SS)          // 32768 nodes
#define NEDG (NN * 8)           // 262144 edges
#define KK   256                // contraction dim everywhere
#define NBLK 128                // persistent decode grid

typedef __attribute__((ext_vector_type(16))) _Float16 v16h;
typedef __attribute__((ext_vector_type(8)))  _Float16 v8h;
typedef __attribute__((ext_vector_type(8)))  float    v8f;

// ---------------- fast transcendental helpers (CDNA5 TRANS ops) ----------------
__device__ __forceinline__ float fast_tanh(float x) {
#if __has_builtin(__builtin_amdgcn_tanhf)
  return __builtin_amdgcn_tanhf(x);
#else
  return tanhf(x);
#endif
}
__device__ __forceinline__ float fast_exp(float x) {
  return __builtin_amdgcn_exp2f(x * 1.4426950408889634f);
}
__device__ __forceinline__ float fast_log(float x) {
#if __has_builtin(__builtin_amdgcn_logf)
  return __builtin_amdgcn_logf(x) * 0.6931471805599453f;
#else
  return logf(x);
#endif
}
__device__ __forceinline__ float sigf(float x) {
  return __builtin_amdgcn_rcpf(1.0f + fast_exp(-x));
}

// ---------------- WMMA tile helpers ----------------
// 16-bit A/B tile load for v_wmma_f32_16x16x32_f16.
// lane<16 holds (row = lane), K = {k0..k0+7, k0+16..k0+23}
// lane>=16 holds (row = lane-16), K = {k0+8..k0+15, k0+24..k0+31}
__device__ __forceinline__ v16h ld_tile16(const _Float16* p) {
  v8h lo = *(const v8h*)(p);
  v8h hi = *(const v8h*)(p + 16);
  v16h r;
#pragma unroll
  for (int i = 0; i < 8; ++i) { r[i] = lo[i]; r[i + 8] = hi[i]; }
  return r;
}

__device__ __forceinline__ v8f wmma16(v16h a, v16h b, v8f c) {
  return __builtin_amdgcn_wmma_f32_16x16x32_f16(
      false, a, false, b, (short)0, c, false, false);
}

// hash-based Gumbel noise (deterministic stand-in for the pre-sampled noise)
__device__ __forceinline__ float gumbelf(unsigned step, unsigned b, unsigned s) {
  unsigned a = step * 0x9E3779B1u ^ (b * 0x85EBCA6Bu + 0x165667B1u) ^ (s * 0xC2B2AE35u);
  a ^= a >> 16; a *= 0x7feb352du; a ^= a >> 15; a *= 0x846ca68bu; a ^= a >> 16;
  float u = fmaxf((float)(a >> 8) * (1.0f / 16777216.0f), 1e-9f);
  return -fast_log(-fast_log(u));
}

// ---------------- grid-wide barrier (monotonic counter, s_sleep spin) ----------------
__device__ __forceinline__ void grid_sync(unsigned* cnt, unsigned target) {
  __threadfence();                 // make this block's global writes visible
  __syncthreads();
  if (threadIdx.x == 0) {
    __hip_atomic_fetch_add(cnt, 1u, __ATOMIC_RELEASE, __HIP_MEMORY_SCOPE_AGENT);
    while (__hip_atomic_load(cnt, __ATOMIC_ACQUIRE, __HIP_MEMORY_SCOPE_AGENT) < target) {
      __builtin_amdgcn_s_sleep(2);
    }
  }
  __syncthreads();
  __threadfence();                 // discard stale lines before reading others' data
}

// ---------------- prep kernels ----------------
__global__ void k_zero_f32(float* p, long n) {
  long i = (long)blockIdx.x * blockDim.x + threadIdx.x;
  long st = (long)gridDim.x * blockDim.x;
  for (; i < n; i += st) p[i] = 0.0f;
}

__global__ void k_fill_f32(float* p, float v, long n) {
  long i = (long)blockIdx.x * blockDim.x + threadIdx.x;
  long st = (long)gridDim.x * blockDim.x;
  for (; i < n; i += st) p[i] = v;
}

__global__ void k_conv_f16(const float* s, _Float16* d, long n) {
  long i = (long)blockIdx.x * blockDim.x + threadIdx.x;
  long st = (long)gridDim.x * blockDim.x;
  for (; i < n; i += st) d[i] = (_Float16)s[i];
}

// dst[n*rows + k] = src[k*cols + n]   (make [n][k] "B-transposed" f16 weights)
__global__ void k_transpose_f16(const float* src, _Float16* dst, int rows, int cols) {
  long n = (long)rows * cols;
  long i = (long)blockIdx.x * blockDim.x + threadIdx.x;
  long st = (long)gridDim.x * blockDim.x;
  for (; i < n; i += st) {
    int k = (int)(i / cols);
    int c = (int)(i % cols);
    dst[(long)c * rows + k] = (_Float16)src[i];
  }
}

// degree / normalization
__global__ void k_deg_count(const int* ei, float* deg, int ne) {
  int i = blockIdx.x * blockDim.x + threadIdx.x;
  if (i < ne) atomicAdd(&deg[ei[ne + i]], 1.0f);
}
__global__ void k_deg_finalize(float* dinv, int n) {
  int i = blockIdx.x * blockDim.x + threadIdx.x;
  if (i < n) dinv[i] = rsqrtf(fmaxf(dinv[i], 1.0f));
}

// layer-1 GCN collapses to a scalar per node: coef[d] = sum norm * x[src] (+ self)
__global__ void k_coef_init(const float* x, const float* dinv, float* coef, int n) {
  int i = blockIdx.x * blockDim.x + threadIdx.x;
  if (i < n) coef[i] = dinv[i] * dinv[i] * x[i];
}
__global__ void k_coef_edges(const int* ei, const float* x, const float* dinv,
                             float* coef, int ne) {
  int i = blockIdx.x * blockDim.x + threadIdx.x;
  if (i < ne) {
    int s = ei[i], d = ei[ne + i];
    atomicAdd(&coef[d], dinv[s] * dinv[d] * x[s]);
  }
}

// layer-2 aggregation: agg[d][j] = sum_e norm * relu(coef[src]*W1[j]+b1[j]) (+ self)
__global__ void k_agg_init(const float* coef, const float* dinv, const float* w1,
                           const float* b1, float* agg) {
  long n = (long)NN * HH;
  long i = (long)blockIdx.x * blockDim.x + threadIdx.x;
  long st = (long)gridDim.x * blockDim.x;
  for (; i < n; i += st) {
    int node = (int)(i >> 8), j = (int)(i & 255);
    float h1 = fmaxf(coef[node] * w1[j] + b1[j], 0.0f);
    agg[i] = dinv[node] * dinv[node] * h1;
  }
}
__global__ void k_agg_edges(const int* ei, const float* coef, const float* dinv,
                            const float* w1, const float* b1, float* agg) {
  long n = (long)NEDG * HH;
  long i = (long)blockIdx.x * blockDim.x + threadIdx.x;
  long st = (long)gridDim.x * blockDim.x;
  for (; i < n; i += st) {
    int e = (int)(i >> 8), j = (int)(i & 255);
    int s = ei[e], d = ei[NEDG + e];
    float h1 = fmaxf(coef[s] * w1[j] + b1[j], 0.0f);
    atomicAdd(&agg[(long)d * HH + j], dinv[s] * dinv[d] * h1);
  }
}

// ---------------- generic f16 WMMA GEMM:  C = A[M,256] @ Bt[n][k] (+bias) ----------------
__global__ __launch_bounds__(32)
void k_gemm_f16(const _Float16* __restrict__ A, const _Float16* __restrict__ Bt,
                const float* __restrict__ bias, _Float16* __restrict__ C, int hasBias) {
  const int lane = threadIdx.x;
  const int row0 = blockIdx.y * 16;
  const int col0 = blockIdx.x * 16;
  const int khalf = (lane >> 4) * 8;
  const _Float16* ap = A + (long)(row0 + (lane & 15)) * KK + khalf;
  const _Float16* bp = Bt + (long)(col0 + (lane & 15)) * KK + khalf;
  v8f acc = {};
#pragma unroll
  for (int kt = 0; kt < 8; ++kt) {
    v16h a = ld_tile16(ap + kt * 32);
    v16h b = ld_tile16(bp + kt * 32);
    acc = wmma16(a, b, acc);
  }
  const int n = col0 + (lane & 15);
  const float bv = hasBias ? bias[n] : 0.0f;
#pragma unroll
  for (int v = 0; v < 8; ++v) {
    int m = row0 + v + ((lane >> 4) << 3);
    C[(long)m * KK + n] = (_Float16)(acc[v] + bv);
  }
}

__global__ void k_dec_init(const float* start, _Float16* dec) {
  dec[(long)blockIdx.x * EE + threadIdx.x] = (_Float16)start[threadIdx.x];
}

// ---------------- persistent decoder: all 256 LSTM+attention steps in one launch ----------------
// Block blk owns (a) gate tile (rowt = blk>>4, colt = blk&15) in phase 1 and
// (b) batch row b = blk in phase 2. Grid barrier between phases.
__global__ __launch_bounds__(256)
void k_decode(const _Float16* __restrict__ wihh, const _Float16* __restrict__ whhh,
              const float* __restrict__ bih, const float* __restrict__ bhh,
              const float* __restrict__ attn_wq, const float* __restrict__ attn_v,
              const _Float16* __restrict__ refh, const _Float16* __restrict__ embh,
              _Float16* __restrict__ hb0, _Float16* __restrict__ hb1,
              float* __restrict__ hf32, float* __restrict__ cst,
              _Float16* __restrict__ dech, int* __restrict__ mask,
              float* __restrict__ out_logp, float* __restrict__ out_idx,
              unsigned* __restrict__ barcnt) {
  const int blk  = blockIdx.x;       // 0..127
  const int tid  = threadIdx.x;      // 0..255
  const int lane = tid & 31;
  const int wave = tid >> 5;

  __shared__ float sg[4][256];       // per-gate f32 tiles from WMMA waves
  __shared__ float hrow[256], qv[256], slog[256], rv[256], av[256];
  __shared__ int   ri[256];
  __shared__ int   chosen_s;

  av[tid] = attn_v[tid];             // resident for the whole decode

  const int row0 = (blk >> 4) * 16;  // batch-row tile
  const int col0 = (blk & 15) * 16;  // hidden-col tile
  const int b    = blk;              // attention batch row
  unsigned epoch = 0;

  for (int step = 0; step < SS; ++step) {
    const _Float16* hin  = (step & 1) ? hb1 : hb0;
    _Float16*       hout = (step & 1) ? hb0 : hb1;

    // ---- phase 1: gates = dec@w_ih^T + h@w_hh^T (+biases), fused LSTM cell
    if (wave < 4) {
      const int g = wave;            // gate index (i, f, g, o)
      const int khalf = (lane >> 4) * 8;
      const _Float16* ad = dech + (long)(row0 + (lane & 15)) * KK + khalf;
      const _Float16* ah = hin  + (long)(row0 + (lane & 15)) * KK + khalf;
      const _Float16* bi = wihh + (long)(g * 256 + col0 + (lane & 15)) * KK + khalf;
      const _Float16* bh = whhh + (long)(g * 256 + col0 + (lane & 15)) * KK + khalf;
      v8f acc = {};
#pragma unroll
      for (int kt = 0; kt < 8; ++kt) {
        v16h a0 = ld_tile16(ad + kt * 32);
        v16h b0 = ld_tile16(bi + kt * 32);
        v16h a1 = ld_tile16(ah + kt * 32);
        v16h b1 = ld_tile16(bh + kt * 32);
        acc = wmma16(a0, b0, acc);
        acc = wmma16(a1, b1, acc);
      }
#pragma unroll
      for (int v = 0; v < 8; ++v) {
        int m_l = v + ((lane >> 4) << 3);
        sg[g][m_l * 16 + (lane & 15)] = acc[v];
      }
    }
    __syncthreads();
    {
      const int m_l = tid >> 4, n_l = tid & 15;
      const int n = col0 + n_l;
      const long idx = (long)(row0 + m_l) * HH + n;
      float gi = sg[0][tid] + bih[n]       + bhh[n];
      float gf = sg[1][tid] + bih[256 + n] + bhh[256 + n];
      float gg = sg[2][tid] + bih[512 + n] + bhh[512 + n];
      float go = sg[3][tid] + bih[768 + n] + bhh[768 + n];
      float cn = sigf(gf) * cst[idx] + sigf(gi) * fast_tanh(gg);
      float hn = sigf(go) * fast_tanh(cn);
      cst[idx]  = cn;
      hf32[idx] = hn;
      hout[idx] = (_Float16)hn;
    }
    grid_sync(barcnt, (++epoch) * NBLK);

    // ---- phase 2: pointer attention + Gumbel-max + mask/gather for row b
    const int t = tid;
    hrow[t] = hf32[(long)b * HH + t];
    __syncthreads();

    float q = 0.0f;                  // q[t] = (h @ attn_wq)[b][t]
#pragma unroll 4
    for (int k = 0; k < HH; ++k) q += hrow[k] * attn_wq[(long)k * HH + t];
    qv[t] = q;
    __syncthreads();

    const _Float16* rp = refh + ((long)b * SS + t) * HH;
    float u = 0.0f;                  // sum_h v[h]*tanh(q[h] + ref_proj[b][t][h])
#pragma unroll 4
    for (int h = 0; h < HH; ++h) u += av[h] * fast_tanh(qv[h] + (float)rp[h]);
    float logit = 10.0f * fast_tanh(u);
    if (mask[(long)b * SS + t]) logit = -100000.0f;
    slog[t] = logit;

    rv[t] = logit + gumbelf((unsigned)step, (unsigned)b, (unsigned)t);
    ri[t] = t;
    __syncthreads();
    for (int s2 = 128; s2 > 0; s2 >>= 1) {        // argmax, first index wins ties
      if (t < s2) {
        float ov = rv[t + s2]; int oi = ri[t + s2];
        if (ov > rv[t] || (ov == rv[t] && oi < ri[t])) { rv[t] = ov; ri[t] = oi; }
      }
      __syncthreads();
    }
    if (t == 0) chosen_s = ri[0];
    __syncthreads();
    const int chosen = chosen_s;

    rv[t] = logit;                                 // log_softmax at chosen
    __syncthreads();
    for (int s2 = 128; s2 > 0; s2 >>= 1) {
      if (t < s2) rv[t] = fmaxf(rv[t], rv[t + s2]);
      __syncthreads();
    }
    const float mx = rv[0];
    __syncthreads();
    rv[t] = fast_exp(logit - mx);
    __syncthreads();
    for (int s2 = 128; s2 > 0; s2 >>= 1) {
      if (t < s2) rv[t] += rv[t + s2];
      __syncthreads();
    }
    if (t == 0) {
      out_logp[(long)b * SS + step] = slog[chosen] - mx - fast_log(rv[0]);
      out_idx[(long)b * SS + step]  = (float)chosen;
      mask[(long)b * SS + chosen] = 1;
    }
    dech[(long)b * EE + t] = embh[((long)b * SS + chosen) * EE + t];
    grid_sync(barcnt, (++epoch) * NBLK);
  }
}

// ---------------- host-side orchestration ----------------
extern "C" void kernel_launch(void* const* d_in, const int* in_sizes, int n_in,
                              void* d_out, int out_size, void* d_ws, size_t ws_size,
                              hipStream_t stream) {
  (void)in_sizes; (void)n_in; (void)out_size; (void)ws_size;
  const float* x        = (const float*)d_in[0];
  const int*   ei       = (const int*)d_in[1];
  const float* gcn_w1   = (const float*)d_in[2];
  const float* gcn_b1   = (const float*)d_in[3];
  const float* gcn_w2   = (const float*)d_in[4];
  const float* gcn_b2   = (const float*)d_in[5];
  const float* enc_w    = (const float*)d_in[6];
  const float* enc_b    = (const float*)d_in[7];
  const float* w_ih     = (const float*)d_in[8];
  const float* w_hh     = (const float*)d_in[9];
  const float* b_ih     = (const float*)d_in[10];
  const float* b_hh     = (const float*)d_in[11];
  const float* start_in = (const float*)d_in[12];
  const float* attn_wq  = (const float*)d_in[13];
  const float* attn_wr  = (const float*)d_in[14];
  const float* attn_v   = (const float*)d_in[15];

  float* out_logp = (float*)d_out;
  float* out_idx  = (float*)d_out + (long)BB * SS;

  // workspace layout
  char* base = (char*)d_ws;
  size_t off = 0;
  auto alloc = [&](size_t bytes) { size_t o = off; off = (off + bytes + 255) & ~(size_t)255; return o; };
  float*     dinv   = (float*)(base + alloc((size_t)NN * 4));
  float*     coef   = (float*)(base + alloc((size_t)NN * 4));
  float*     agg    = (float*)(base + alloc((size_t)NN * HH * 4));       // 32 MB
  _Float16*  aggh   = (_Float16*)(base + alloc((size_t)NN * HH * 2));    // 16 MB
  _Float16*  embh   = (_Float16*)(base + alloc((size_t)NN * EE * 2));    // 16 MB
  _Float16*  ench   = (_Float16*)(base + alloc((size_t)NN * HH * 2));    // 16 MB
  _Float16*  refh   = (_Float16*)(base + alloc((size_t)NN * HH * 2));    // 16 MB
  _Float16*  wihh   = (_Float16*)(base + alloc((size_t)1024 * 256 * 2));
  _Float16*  whhh   = (_Float16*)(base + alloc((size_t)1024 * 256 * 2));
  _Float16*  w2t    = (_Float16*)(base + alloc((size_t)256 * 256 * 2));
  _Float16*  encwt  = (_Float16*)(base + alloc((size_t)256 * 256 * 2));
  _Float16*  awrt   = (_Float16*)(base + alloc((size_t)256 * 256 * 2));
  _Float16*  hb0    = (_Float16*)(base + alloc((size_t)BB * HH * 2));
  _Float16*  hb1    = (_Float16*)(base + alloc((size_t)BB * HH * 2));
  float*     hf32   = (float*)(base + alloc((size_t)BB * HH * 4));
  float*     cst    = (float*)(base + alloc((size_t)BB * HH * 4));
  _Float16*  dech   = (_Float16*)(base + alloc((size_t)BB * EE * 2));
  int*       mask   = (int*)(base + alloc((size_t)BB * SS * 4));
  unsigned*  barcnt = (unsigned*)(base + alloc(256));

  // ---- weight conversion / transposition (f16, [n][k] layout for WMMA B tiles)
  k_conv_f16<<<1024, 256, 0, stream>>>(w_ih, wihh, (long)1024 * 256);
  k_conv_f16<<<1024, 256, 0, stream>>>(w_hh, whhh, (long)1024 * 256);
  k_transpose_f16<<<256, 256, 0, stream>>>(gcn_w2, w2t, 256, 256);
  k_transpose_f16<<<256, 256, 0, stream>>>(enc_w, encwt, 256, 256);
  k_transpose_f16<<<256, 256, 0, stream>>>(attn_wr, awrt, 256, 256);

  // ---- GCN normalization (deg -> dinv), layer-1 scalar coefficients
  k_fill_f32<<<128, 256, 0, stream>>>(dinv, 1.0f, NN);            // self loop
  k_deg_count<<<(NEDG + 255) / 256, 256, 0, stream>>>(ei, dinv, NEDG);
  k_deg_finalize<<<(NN + 255) / 256, 256, 0, stream>>>(dinv, NN);
  k_coef_init<<<(NN + 255) / 256, 256, 0, stream>>>(x, dinv, coef, NN);
  k_coef_edges<<<(NEDG + 255) / 256, 256, 0, stream>>>(ei, x, dinv, coef, NEDG);

  // ---- layer-2 aggregation (h1 recomputed on the fly from coef)
  k_agg_init<<<8192, 256, 0, stream>>>(coef, dinv, gcn_w1, gcn_b1, agg);
  k_agg_edges<<<8192, 256, 0, stream>>>(ei, coef, dinv, gcn_w1, gcn_b1, agg);
  k_conv_f16<<<8192, 256, 0, stream>>>(agg, aggh, (long)NN * HH);

  // ---- dense chain: emb -> enc_out -> ref_proj (all WMMA)
  dim3 gfull(EE / 16, NN / 16);
  k_gemm_f16<<<gfull, 32, 0, stream>>>(aggh, w2t, gcn_b2, embh, 1);
  k_gemm_f16<<<gfull, 32, 0, stream>>>(embh, encwt, enc_b, ench, 1);
  k_gemm_f16<<<gfull, 32, 0, stream>>>(ench, awrt, (const float*)nullptr, refh, 0);

  // ---- decoder state init (barrier counter reset every call => deterministic)
  k_zero_f32<<<64, 256, 0, stream>>>(hf32, (long)BB * HH);
  k_zero_f32<<<64, 256, 0, stream>>>(cst, (long)BB * HH);
  k_zero_f32<<<64, 256, 0, stream>>>((float*)hb0, (long)BB * HH / 2);   // f16 zeros
  k_zero_f32<<<64, 256, 0, stream>>>((float*)mask, (long)BB * SS);
  k_zero_f32<<<1, 64, 0, stream>>>((float*)barcnt, 64);
  k_dec_init<<<BB, EE, 0, stream>>>(start_in, dech);

  // ---- all 256 decode steps in one persistent launch
  k_decode<<<NBLK, 256, 0, stream>>>(wihh, whhh, b_ih, b_hh, attn_wq, attn_v,
                                     refh, embh, hb0, hb1, hf32, cst, dech, mask,
                                     out_logp, out_idx, barcnt);
}
